// MixFusionFeedForward_72971494359719
// MI455X (gfx1250) — compile-verified, matrix operand order checked by reading, not run on hardware
//
#include <hip/hip_runtime.h>
#include <hip/hip_bf16.h>

// ---------- types ----------
typedef __attribute__((ext_vector_type(16))) __bf16 v16bf;
typedef __attribute__((ext_vector_type(8)))  float  v8f;
typedef __attribute__((ext_vector_type(8)))  __bf16 bf16x8;
typedef __attribute__((ext_vector_type(4)))  __bf16 bf16x4;
typedef unsigned int u32x4 __attribute__((ext_vector_type(4)));
typedef int          i32x8 __attribute__((ext_vector_type(8)));
typedef int          i32x4 __attribute__((ext_vector_type(4)));

// Problem constants
#define MROWS 11520           // 2*5760 = 16*720
#define K1    512
#define N1    1960
#define N1P   2048            // w1t rows padded (zero rows 1960..2047)
#define K2P   1984            // 1960 padded to 62*32
#define K2V   1960
#define N2    512
#define LDSP  40              // LDS row stride (bf16 elems): 32 + 8 pad

// ---- Tensor Data Mover (6-arg builtin on this toolchain) ----
#if __has_builtin(__builtin_amdgcn_tensor_load_to_lds) && \
    __has_builtin(__builtin_amdgcn_s_wait_tensorcnt)
#define HAVE_TDM 1
#else
#define HAVE_TDM 0
#endif

#if HAVE_TDM
// DMA a 128-row x 32-bf16 tile (row stride = row_stride_elems) from global
// into LDS at lds_addr, with hardware padding of 4 DWORDs after each 64B row
// (reproduces the LDSP=40 layout).
__device__ __forceinline__ void tdm_load_tile_128x32(
    unsigned lds_addr, const __bf16* gsrc, int row_stride_elems) {
  unsigned long long ga = (unsigned long long)(size_t)gsrc;
  u32x4 g0;
  g0[0] = 1u;                                   // count=1 (valid user D#)
  g0[1] = lds_addr;                             // lds_addr (bytes)
  g0[2] = (unsigned)(ga & 0xFFFFFFFFu);         // global_addr[31:0]
  g0[3] = (unsigned)((ga >> 32) & 0x01FFFFFFu)  // global_addr[56:32]
        | (2u << 30);                           // type = 2 ("image")
  i32x8 g1;
  g1[0] = (1 << 16)                             // data_size = 2 bytes
        | (1 << 20)                             // pad_enable
        | (3 << 22)                             // pad_interval: 16 DWORDs (one 64B row)
        | (3 << 25);                            // pad_amount: 4 DWORDs (8 bf16)
  g1[1] = (32 << 16);                           // tensor_dim0 lo = 32
  g1[2] = (128 << 16);                          // tensor_dim0 hi | tensor_dim1 lo = 128
  g1[3] = (32 << 16);                           // tensor_dim1 hi | tile_dim0 = 32
  g1[4] = 128;                                  // tile_dim1 = 128 | tile_dim2 = 0
  g1[5] = row_stride_elems;                     // tensor_dim0_stride lo (elements)
  g1[6] = 0;                                    // stride0 hi | stride1 lo
  g1[7] = 0;                                    // stride1 hi
  i32x4 z4 = {0, 0, 0, 0};
  i32x8 z8 = {0, 0, 0, 0, 0, 0, 0, 0};
  __builtin_amdgcn_tensor_load_to_lds(g0, g1, z4, z4, z8, 0);
}
#endif

union FragCvt {
  struct P { bf16x8 lo, hi; } p;
  v16bf b;
};

// Load one 16x32 bf16 fragment (A or B^T) from an LDS tile row (stride LDSP).
// lane m = lane&15 selects the row; half = lane>>4 selects K sub-ranges
// [half*8, half*8+8) and [16+half*8, 16+half*8+8)  (two ds_load_b128).
__device__ __forceinline__ v16bf load_frag(const __bf16* rowp, int hh) {
  FragCvt f;
  f.p.lo = *(const bf16x8*)(rowp + hh * 8);
  f.p.hi = *(const bf16x8*)(rowp + 16 + hh * 8);
  return f.b;
}

// ---------------------------------------------------------------------------
// Prep kernels (run once per launch):
// xb [11520][512]  : bf16 copy of x (removes per-tile f32 reads + cvt in GEMM1)
// w1t[2048][512]   : w1t[n][k] = w1[k][n]  (rows >=1960 zero)
// w2t[512][1984]   : w2t[n][k] = w2[k][n]  (cols >=1960 zero)
// ---------------------------------------------------------------------------
__global__ __launch_bounds__(256) void prep_xb_kernel(
    const float* __restrict__ x, __bf16* __restrict__ xb) {
  int idx = blockIdx.x * 256 + threadIdx.x;    // float4 slots
  if (idx >= (MROWS * K1) / 4) return;
  const float4 v = *(const float4*)(x + (size_t)idx * 4);
  bf16x4 h = {(__bf16)v.x, (__bf16)v.y, (__bf16)v.z, (__bf16)v.w};
  *(bf16x4*)(xb + (size_t)idx * 4) = h;
}
__global__ __launch_bounds__(256) void prep_w1t_kernel(
    const float* __restrict__ w1, __bf16* __restrict__ w1t) {
  int idx = blockIdx.x * 256 + threadIdx.x;
  if (idx >= N1P * K1) return;
  int k = idx & (K1 - 1), n = idx >> 9;
  w1t[idx] = (n < N1) ? (__bf16)w1[(size_t)k * N1 + n] : (__bf16)0.0f;
}
__global__ __launch_bounds__(256) void prep_w2t_kernel(
    const float* __restrict__ w2, __bf16* __restrict__ w2t) {
  int idx = blockIdx.x * 256 + threadIdx.x;
  if (idx >= N2 * K2P) return;
  int k = idx % K2P, n = idx / K2P;
  w2t[idx] = (k < K2V) ? (__bf16)w2[(size_t)k * N2 + n] : (__bf16)0.0f;
}

// ---------------------------------------------------------------------------
// Unified bf16 WMMA GEMM: out[M][ldOut] = Amat[M][KDIM] @ Bt[N][KDIM]^T + bias
// 256 threads = 8 waves; block tile 128(M) x 128(N); wave tile 32x64;
// double-buffered LDS; both operands staged by the Tensor Data Mover.
// NGUARD: guard output columns against NMAX (for N=1960 edge tiles).
// ---------------------------------------------------------------------------
template <int KDIM, bool NGUARD, int NMAX>
__global__ __launch_bounds__(256) void gemm_kernel(
    const __bf16* __restrict__ Amat, const __bf16* __restrict__ Bt,
    const float* __restrict__ bias, float* __restrict__ out, int ldOut) {
  __shared__ __bf16 As[2][128 * LDSP];
  __shared__ __bf16 Bs[2][128 * LDSP];

  const int tid   = threadIdx.x;
  const int m0    = blockIdx.y * 128;
  const int n0    = blockIdx.x * 128;
  const int wave  = tid >> 5;
  const int lane  = tid & 31;
  const int hh    = lane >> 4;
  const int ml    = lane & 15;
  const int mbase = (wave & 3) * 32;   // 4 waves along M
  const int nbase = (wave >> 2) * 64;  // 2 waves along N

  v8f acc[2][4] = {};

  auto stage = [&](int buf, int k0) {
#if HAVE_TDM
    if (tid < 32) {  // one wave issues both DMAs (TDM ignores EXEC)
      tdm_load_tile_128x32((unsigned)(size_t)&As[buf][0],
                           Amat + (size_t)m0 * KDIM + k0, KDIM);
      tdm_load_tile_128x32((unsigned)(size_t)&Bs[buf][0],
                           Bt + (size_t)n0 * KDIM + k0, KDIM);
    }
#else
#pragma unroll
    for (int it = 0; it < 2; ++it) {
      int lin = tid + it * 256;          // 0..511 vec8 slots
      int r = lin >> 2, cv = lin & 3;
      *(bf16x8*)(&As[buf][r * LDSP + cv * 8]) =
          *(const bf16x8*)(Amat + (size_t)(m0 + r) * KDIM + k0 + cv * 8);
      *(bf16x8*)(&Bs[buf][r * LDSP + cv * 8]) =
          *(const bf16x8*)(Bt + (size_t)(n0 + r) * KDIM + k0 + cv * 8);
    }
#endif
  };
  auto tdm_wait = [&]() {
#if HAVE_TDM
    if (tid < 32) __builtin_amdgcn_s_wait_tensorcnt(0);
#endif
  };

  stage(0, 0);
  tdm_wait();
  __syncthreads();

  int cur = 0;
  for (int k0 = 0; k0 < KDIM; k0 += 32) {
    if (k0 + 32 < KDIM) {
      stage(cur ^ 1, k0 + 32);
      __builtin_prefetch(Amat + (size_t)(m0 + (tid >> 1)) * KDIM + k0 + 64, 0, 1);
    }
    v16bf a0 = load_frag(&As[cur][(mbase + ml) * LDSP], hh);
    v16bf a1 = load_frag(&As[cur][(mbase + 16 + ml) * LDSP], hh);
#pragma unroll
    for (int j = 0; j < 4; ++j) {
      v16bf bf = load_frag(&Bs[cur][(nbase + j * 16 + ml) * LDSP], hh);
      acc[0][j] = __builtin_amdgcn_wmma_f32_16x16x32_bf16(false, a0, false, bf, (short)0, acc[0][j], false, false);
      acc[1][j] = __builtin_amdgcn_wmma_f32_16x16x32_bf16(false, a1, false, bf, (short)0, acc[1][j], false, false);
    }
    tdm_wait();
    __syncthreads();
    cur ^= 1;
  }

#pragma unroll
  for (int i = 0; i < 2; ++i)
#pragma unroll
    for (int j = 0; j < 4; ++j) {
      int col = n0 + nbase + j * 16 + ml;
      if (!NGUARD || col < NMAX) {
        float bv = bias[col];
#pragma unroll
        for (int r = 0; r < 8; ++r) {
          int row = m0 + mbase + i * 16 + hh * 8 + r;
          out[(size_t)row * ldOut + col] = acc[i][j][r] + bv;
        }
      }
    }
}

// ---------------------------------------------------------------------------
// Smooth: S[16,40,60,108] = fold(Y)/fold(ones) — gather of <=9 patch taps.
// ---------------------------------------------------------------------------
__global__ __launch_bounds__(256) void smooth_kernel(
    const float* __restrict__ Y, float* __restrict__ S) {
  int idx = blockIdx.x * 256 + threadIdx.x;
  if (idx >= 16 * 40 * 60 * 108) return;
  int x = idx % 108;
  int y = (idx / 108) % 60;
  int c = (idx / (108 * 60)) % 40;
  int t = idx / (108 * 60 * 40);
  int yp = y + 3, xp = x + 3;
  int i0 = (yp >= 6) ? (yp - 4) / 3 : 0;   // ceil((yp-6)/3)
  int i1 = yp / 3; if (i1 > 19) i1 = 19;
  int j0 = (xp >= 6) ? (xp - 4) / 3 : 0;
  int j1 = xp / 3; if (j1 > 35) j1 = 35;
  float sum = 0.f; int cnt = 0;
  for (int i = i0; i <= i1; ++i) {
    int ky = yp - 3 * i;
    for (int j = j0; j <= j1; ++j) {
      int kx = xp - 3 * j;
      int row = t * 720 + i * 36 + j;
      sum += Y[(size_t)row * N1 + c * 49 + ky * 7 + kx];
      ++cnt;
    }
  }
  S[idx] = sum / (float)cnt;
}

// ---------------------------------------------------------------------------
// Fuse: unfold(S) -> depthwise conv (3x3 / 5x5 on patch grid) -> exact GELU
// -> bf16 G[11520][1984] (cols 1960..1983 zero-padded for GEMM2's K loop).
// ---------------------------------------------------------------------------
__global__ __launch_bounds__(256) void fuse_kernel(
    const float* __restrict__ S,
    const float* __restrict__ w3, const float* __restrict__ b3,
    const float* __restrict__ w5, const float* __restrict__ b5,
    __bf16* __restrict__ G) {
  int idx = blockIdx.x * 256 + threadIdx.x;
  if (idx >= MROWS * K2P) return;
  int ch = idx % K2P;
  int row = idx / K2P;
  if (ch >= K2V) { G[idx] = (__bf16)0.0f; return; }

  int t = row / 720, l = row % 720;
  int i = l / 36, j = l % 36;
  int c = ch / 49, k = ch % 49;
  int ky = k / 7, kx = k % 7;
  const float* Sc = S + (size_t)(t * 40 + c) * 60 * 108;
  int y0 = i * 3 + ky - 3;
  int x0 = j * 3 + kx - 3;

  float acc;
  if (ch < 980) {
    const float* w = w3 + ch * 9;
    acc = b3[ch];
#pragma unroll
    for (int di = -1; di <= 1; ++di) {
      int ii = i + di; if (ii < 0 || ii >= 20) continue;
      int yy = y0 + 3 * di;
      if (yy < 0 || yy >= 60) continue;
#pragma unroll
      for (int dj = -1; dj <= 1; ++dj) {
        int jj = j + dj; if (jj < 0 || jj >= 36) continue;
        int xx = x0 + 3 * dj;
        if (xx < 0 || xx >= 108) continue;
        acc += w[(di + 1) * 3 + (dj + 1)] * Sc[yy * 108 + xx];
      }
    }
  } else {
    const float* w = w5 + (ch - 980) * 25;
    acc = b5[ch - 980];
#pragma unroll
    for (int di = -2; di <= 2; ++di) {
      int ii = i + di; if (ii < 0 || ii >= 20) continue;
      int yy = y0 + 3 * di;
      if (yy < 0 || yy >= 60) continue;
#pragma unroll
      for (int dj = -2; dj <= 2; ++dj) {
        int jj = j + dj; if (jj < 0 || jj >= 36) continue;
        int xx = x0 + 3 * dj;
        if (xx < 0 || xx >= 108) continue;
        acc += w[(di + 2) * 5 + (dj + 2)] * Sc[yy * 108 + xx];
      }
    }
  }
  float g = 0.5f * acc * (1.0f + erff(acc * 0.70710678118654752f));
  G[idx] = (__bf16)g;
}

// ---------------------------------------------------------------------------
extern "C" void kernel_launch(void* const* d_in, const int* in_sizes, int n_in,
                              void* d_out, int out_size, void* d_ws, size_t ws_size,
                              hipStream_t stream) {
  const float* x  = (const float*)d_in[0];   // (2,5760,512)
  const float* w1 = (const float*)d_in[1];   // (512,1960)
  const float* b1 = (const float*)d_in[2];   // (1960,)
  const float* w3 = (const float*)d_in[3];   // (980,1,3,3)
  const float* b3 = (const float*)d_in[4];   // (980,)
  const float* w5 = (const float*)d_in[5];   // (980,1,5,5)
  const float* b5 = (const float*)d_in[6];   // (980,)
  const float* w2 = (const float*)d_in[7];   // (1960,512)
  const float* b2 = (const float*)d_in[8];   // (512,)
  float* out = (float*)d_out;                // (2,5760,512)

  // Workspace layout
  float*  Y   = (float*)d_ws;                                 // 11520*1960 f32
  float*  S   = Y + (size_t)MROWS * N1;                       // 16*40*60*108 f32
  __bf16* G   = (__bf16*)(S + (size_t)16 * 40 * 60 * 108);    // 11520*1984 bf16
  __bf16* w1t = G + (size_t)MROWS * K2P;                      // 2048*512 bf16
  __bf16* w2t = w1t + (size_t)N1P * K1;                       // 512*1984 bf16
  __bf16* xb  = w2t + (size_t)N2 * K2P;                       // 11520*512 bf16

  // 0) prep: x -> bf16; weights -> transposed bf16 (+ zero pad)
  prep_xb_kernel<<<((MROWS * K1) / 4 + 255) / 256, 256, 0, stream>>>(x, xb);
  prep_w1t_kernel<<<(N1P * K1 + 255) / 256, 256, 0, stream>>>(w1, w1t);
  prep_w2t_kernel<<<(N2 * K2P + 255) / 256, 256, 0, stream>>>(w2, w2t);

  // 1) GEMM1 (bf16 WMMA, TDM A+B staging): Y = x @ w1 + b1
  gemm_kernel<K1, true, N1>
      <<<dim3((N1 + 127) / 128, MROWS / 128), 256, 0, stream>>>(xb, w1t, b1, Y, N1);

  // 2) fold/unfold smoothing gather
  smooth_kernel<<<(16 * 40 * 60 * 108 + 255) / 256, 256, 0, stream>>>(Y, S);

  // 3) unfold + depthwise conv + GELU -> bf16 (K padded to 1984)
  fuse_kernel<<<(MROWS * K2P + 255) / 256, 256, 0, stream>>>(S, w3, b3, w5, b5, G);

  // 4) GEMM2 (bf16 WMMA, TDM A+B staging): out = gelu @ w2 + b2
  gemm_kernel<K2P, false, N2>
      <<<dim3(N2 / 128, MROWS / 128), 256, 0, stream>>>(G, w2t, b2, out, N2);
}
